// CausalGIN_8340826488978
// MI455X (gfx1250) — matrix operand order repaired and checked by reading, not compiled, not run on hardware
//
#include <hip/hip_runtime.h>
#include <math.h>

#define N_NODES 50000
#define N_EDGES 400000
#define HDIM    128
#define NCLS    10
#define NLAYERS 3
#define NGRAPH  500
#define BN_EPS  1e-5f

// padded LDS strides (in halves): 136 halves = 272 B = 68 dwords -> stride-4 banks
#define SWPAD 136
#define SAPAD 136

typedef __attribute__((ext_vector_type(16))) _Float16 v16h;
typedef __attribute__((ext_vector_type(8)))  _Float16 v8h;
typedef __attribute__((ext_vector_type(8)))  float    v8f;

__device__ __forceinline__ void atomAdd(float* p, float v) {
    unsafeAtomicAdd(p, v);   // -> global_atomic_add_f32 on gfx1250
}

// ---------------------------------------------------------------- utilities
__global__ void zero_kernel(float* __restrict__ p, long n) {
    long i = (long)blockIdx.x * blockDim.x + threadIdx.x;
    long st = (long)gridDim.x * blockDim.x;
    for (; i < n; i += st) p[i] = 0.f;
}

__global__ void copy_kernel(float* __restrict__ dst, const float* __restrict__ src, long n) {
    long i = (long)blockIdx.x * blockDim.x + threadIdx.x;
    long st = (long)gridDim.x * blockDim.x;
    for (; i < n; i += st) dst[i] = src[i];
}

// ------------------------------------------------------------- batch norm
// blockDim.x == 128; column = tid + 128*blockIdx.y ; stats = [sum(F), sumsq(F)]
__global__ void bn_stats_kernel(const float* __restrict__ v, int rows, int F,
                                float* __restrict__ stats) {
    int col = threadIdx.x + (blockIdx.y << 7);
    float s = 0.f, s2 = 0.f;
    for (int r = blockIdx.x; r < rows; r += gridDim.x) {
        float x = v[(size_t)r * F + col];
        s += x; s2 += x * x;
    }
    atomAdd(&stats[col], s);
    atomAdd(&stats[F + col], s2);
}

template<bool RELU>
__global__ void bn_apply_kernel(const float* __restrict__ v, float* __restrict__ out,
                                int rows, int F, const float* __restrict__ stats,
                                const float* __restrict__ g, const float* __restrict__ b) {
    long n = (long)rows * F;
    long i = (long)blockIdx.x * blockDim.x + threadIdx.x;
    long st = (long)gridDim.x * blockDim.x;
    float inv_rows = 1.f / (float)rows;
    for (; i < n; i += st) {
        int col = (int)(i % F);
        float mean = stats[col] * inv_rows;
        float var  = stats[F + col] * inv_rows - mean * mean;
        float y = (v[i] - mean) * rsqrtf(var + BN_EPS) * g[col] + b[col];
        if (RELU) y = fmaxf(y, 0.f);
        out[i] = y;
    }
}

// ------------------------------------------------------------- WMMA GEMM
// C[numRows x 128] (+)= A[numRows x 128 (lda)] * W[128 x 128 row-major K x N]
// 256 threads = 8 waves; block tile 64 rows x 128 cols; wave w -> cols [16w,16w+16)
// W staged transposed in LDS so each lane's B-fragment halves are contiguous.
// ldc fixed at 128 so stores use immediate offsets.
template<bool ACC, bool BIAS, bool RELU>
__global__ __launch_bounds__(256) void gemm128_kernel(
        const float* __restrict__ A, int lda,
        const float* __restrict__ W,
        const float* __restrict__ bias,
        float* __restrict__ C,
        int numRows)
{
    __shared__ _Float16 sWt[HDIM * SWPAD];   // W transposed: [n][k], padded
    __shared__ _Float16 sA [64   * SAPAD];   // A tile f16, padded
    const int tid  = threadIdx.x;
    const int wave = tid >> 5;
    const int lane = tid & 31;

    for (int i = tid; i < HDIM * HDIM; i += 256) {
        int k = i >> 7, n = i & 127;           // W row-major: i = k*128 + n
        sWt[n * SWPAD + k] = (_Float16)W[i];
    }
    const int row0 = blockIdx.x * 64;
    for (int i = tid; i < 64 * HDIM; i += 256) {
        int r = i >> 7, c = i & 127;
        int gr = row0 + r;
        sA[r * SAPAD + c] = (_Float16)((gr < numRows) ? A[(size_t)gr * lda + c] : 0.f);
    }
    __syncthreads();

    const int colBase = wave << 4;
    const int mrow    = lane & 15;         // A row (in tile) / B,C col (in wave tile)
    const int hi      = lane >> 4;         // 0: lanes 0-15, 1: lanes 16-31
    const int col     = colBase + mrow;
    const bool fullTile = (row0 + 64 <= numRows);
    const float bv = BIAS ? bias[col] : 0.f;

    // B fragments: elem e -> K = k0 + e + 16*hi  => 16 contiguous halves in sWt
    v16h bf[4];
#pragma unroll
    for (int ks = 0; ks < 4; ++ks) {
        const _Float16* bp = &sWt[col * SWPAD + (ks << 5) + (hi << 4)];
        v8h blo = *(const v8h*)(bp);
        v8h bhi = *(const v8h*)(bp + 8);
        bf[ks] = __builtin_shufflevector(blo, bhi,
                                         0, 1, 2, 3, 4, 5, 6, 7,
                                         8, 9, 10, 11, 12, 13, 14, 15);
    }

#pragma unroll
    for (int mt = 0; mt < 4; ++mt) {
        const _Float16* arow = &sA[(mt * 16 + mrow) * SAPAD];
        v8f acc = {0.f, 0.f, 0.f, 0.f, 0.f, 0.f, 0.f, 0.f};
#pragma unroll
        for (int ks = 0; ks < 4; ++ks) {
            const int k0 = ks << 5;
            // A 16x32 f16: elem e -> K = e + 16*hi  => 16 contiguous halves
            v8h alo = *(const v8h*)&arow[k0 + (hi << 4)];
            v8h ahi = *(const v8h*)&arow[k0 + (hi << 4) + 8];
            v16h a = __builtin_shufflevector(alo, ahi,
                                             0, 1, 2, 3, 4, 5, 6, 7,
                                             8, 9, 10, 11, 12, 13, 14, 15);
            acc = __builtin_amdgcn_wmma_f32_16x16x32_f16(
                      false, a, false, bf[ks], (short)0, acc, false, false);
        }
        // C layout: VGPR v -> row = tileRow0 + v (+8 for lanes 16-31), col = colBase+mrow
        const int tileRow0 = row0 + mt * 16 + (hi << 3);
        float* cp = C + (size_t)tileRow0 * HDIM + col;
        if (fullTile) {
#pragma unroll
            for (int v = 0; v < 8; ++v) {
                float val = acc[v];
                if (ACC)  val += cp[v * HDIM];
                if (BIAS) val += bv;
                if (RELU) val = fmaxf(val, 0.f);
                cp[v * HDIM] = val;
            }
        } else {
#pragma unroll
            for (int v = 0; v < 8; ++v) {
                if (tileRow0 + v < numRows) {
                    float val = acc[v];
                    if (ACC)  val += cp[v * HDIM];
                    if (BIAS) val += bv;
                    if (RELU) val = fmaxf(val, 0.f);
                    cp[v * HDIM] = val;
                }
            }
        }
    }
}

// ---------------------------------------------------------- edge / node ops
__global__ void gin_scatter_kernel(const float* __restrict__ h, const int* __restrict__ row,
                                   const int* __restrict__ col, float* __restrict__ z, int E) {
    int t = threadIdx.x;   // 128 threads: one feature each
    for (int e = blockIdx.x; e < E; e += gridDim.x) {
        int r = row[e], c = col[e];
        atomAdd(&z[(size_t)c * HDIM + t], h[(size_t)r * HDIM + t]);
    }
}

__global__ void edge_att_kernel(const float* __restrict__ h, const int* __restrict__ row,
                                const int* __restrict__ col, const float* __restrict__ W,
                                const float* __restrict__ b, float* __restrict__ ewc,
                                float* __restrict__ ewo, int E) {
    __shared__ float s0[128];
    __shared__ float s1[128];
    int t = threadIdx.x;
    for (int e = blockIdx.x; e < E; e += gridDim.x) {
        int r = row[e], c = col[e];
        float hr = h[(size_t)r * HDIM + t];
        float hc = h[(size_t)c * HDIM + t];
        s0[t] = hr * W[2 * t]     + hc * W[2 * (HDIM + t)];
        s1[t] = hr * W[2 * t + 1] + hc * W[2 * (HDIM + t) + 1];
        __syncthreads();
        for (int off = 64; off > 0; off >>= 1) {
            if (t < off) { s0[t] += s0[t + off]; s1[t] += s1[t + off]; }
            __syncthreads();
        }
        if (t == 0) {
            float a0 = s0[0] + b[0], a1 = s1[0] + b[1];
            float m  = fmaxf(a0, a1);
            float e0 = __expf(a0 - m), e1 = __expf(a1 - m);
            float inv = 1.f / (e0 + e1);
            ewc[e] = e0 * inv;
            ewo[e] = e1 * inv;
        }
        __syncthreads();
    }
}

__global__ void node_att_kernel(const float* __restrict__ h, const float* __restrict__ W,
                                const float* __restrict__ b, float* __restrict__ xc,
                                float* __restrict__ xo, int n) {
    __shared__ float s0[128];
    __shared__ float s1[128];
    __shared__ float att[2];
    int t = threadIdx.x;
    for (int i = blockIdx.x; i < n; i += gridDim.x) {
        float hv = h[(size_t)i * HDIM + t];
        s0[t] = hv * W[2 * t];
        s1[t] = hv * W[2 * t + 1];
        __syncthreads();
        for (int off = 64; off > 0; off >>= 1) {
            if (t < off) { s0[t] += s0[t + off]; s1[t] += s1[t + off]; }
            __syncthreads();
        }
        if (t == 0) {
            float a0 = s0[0] + b[0], a1 = s1[0] + b[1];
            float m  = fmaxf(a0, a1);
            float e0 = __expf(a0 - m), e1 = __expf(a1 - m);
            float inv = 1.f / (e0 + e1);
            att[0] = e0 * inv;
            att[1] = e1 * inv;
        }
        __syncthreads();
        xc[(size_t)i * HDIM + t] = att[0] * hv;
        xo[(size_t)i * HDIM + t] = att[1] * hv;
        __syncthreads();
    }
}

__global__ void deg_kernel(const float* __restrict__ ew, const int* __restrict__ row,
                           float* __restrict__ deg, int E) {
    int i = blockIdx.x * blockDim.x + threadIdx.x;
    int st = gridDim.x * blockDim.x;
    for (; i < E; i += st) atomAdd(&deg[row[i]], ew[i]);
}

__global__ void dinv_kernel(float* __restrict__ deg, int n) {
    int i = blockIdx.x * blockDim.x + threadIdx.x;
    int st = gridDim.x * blockDim.x;
    for (; i < n; i += st) {
        float d = deg[i] + 1.0f;   // + self loop weight
        deg[i] = (d > 0.f) ? rsqrtf(fmaxf(d, 1e-12f)) : 0.f;
    }
}

__global__ void selfloop_kernel(const float* __restrict__ vw, const float* __restrict__ dinv,
                                float* __restrict__ out, long n) {
    long i = (long)blockIdx.x * blockDim.x + threadIdx.x;
    long st = (long)gridDim.x * blockDim.x;
    for (; i < n; i += st) {
        float di = dinv[i >> 7];
        out[i] = di * di * vw[i];
    }
}

__global__ void gcn_scatter_kernel(const float* __restrict__ vw, const int* __restrict__ row,
                                   const int* __restrict__ col, const float* __restrict__ ew,
                                   const float* __restrict__ dinv, float* __restrict__ out, int E) {
    int t = threadIdx.x;
    for (int e = blockIdx.x; e < E; e += gridDim.x) {
        int r = row[e], c = col[e];
        float coef = dinv[r] * ew[e] * dinv[c];
        atomAdd(&out[(size_t)c * HDIM + t], coef * vw[(size_t)r * HDIM + t]);
    }
}

__global__ void pool_kernel(const float* __restrict__ v, const int* __restrict__ batch,
                            const float* __restrict__ bias, float* __restrict__ pooled, int n) {
    int t = threadIdx.x;
    for (int i = blockIdx.x; i < n; i += gridDim.x) {
        float val = fmaxf(v[(size_t)i * HDIM + t] + bias[t], 0.f);
        atomAdd(&pooled[(size_t)batch[i] * HDIM + t], val);
    }
}

__global__ void concat_perm_kernel(const float* __restrict__ xcg, const float* __restrict__ xog,
                                   const int* __restrict__ perm, float* __restrict__ out) {
    int total = NGRAPH * 256;
    int i = blockIdx.x * blockDim.x + threadIdx.x;
    int st = gridDim.x * blockDim.x;
    for (; i < total; i += st) {
        int g = i >> 8, t = i & 255;
        out[i] = (t < 128) ? xcg[(size_t)perm[g] * HDIM + t]
                           : xog[(size_t)g * HDIM + (t - 128)];
    }
}

__global__ void fc2_lsm_kernel(const float* __restrict__ v, const float* __restrict__ W,
                               const float* __restrict__ b, float* __restrict__ out, int rows) {
    __shared__ float s[128];
    __shared__ float logits[NCLS];
    __shared__ float red[2];
    int t = threadIdx.x;
    for (int r = blockIdx.x; r < rows; r += gridDim.x) {
        float hv = v[(size_t)r * HDIM + t];
        for (int j = 0; j < NCLS; ++j) {
            s[t] = hv * W[t * NCLS + j];
            __syncthreads();
            for (int off = 64; off > 0; off >>= 1) {
                if (t < off) s[t] += s[t + off];
                __syncthreads();
            }
            if (t == 0) logits[j] = s[0] + b[j];
            __syncthreads();
        }
        if (t == 0) {
            float m = logits[0];
            for (int j = 1; j < NCLS; ++j) m = fmaxf(m, logits[j]);
            float sum = 0.f;
            for (int j = 0; j < NCLS; ++j) sum += __expf(logits[j] - m);
            red[0] = m;
            red[1] = __logf(sum);
        }
        __syncthreads();
        if (t < NCLS) out[(size_t)r * NCLS + t] = logits[t] - red[0] - red[1];
        __syncthreads();
    }
}

// ---------------------------------------------------------------- host side
static void run_bn(const float* v, float* out, int rows, int F, const float* g, const float* b,
                   bool relu, float* stats, hipStream_t stream) {
    zero_kernel<<<8, 256, 0, stream>>>(stats, 2L * F);
    dim3 grid(256, F / 128);
    bn_stats_kernel<<<grid, 128, 0, stream>>>(v, rows, F, stats);
    long n = (long)rows * F;
    int blocks = (int)((n + 255) / 256);
    if (blocks > 4096) blocks = 4096;
    if (relu) bn_apply_kernel<true ><<<blocks, 256, 0, stream>>>(v, out, rows, F, stats, g, b);
    else      bn_apply_kernel<false><<<blocks, 256, 0, stream>>>(v, out, rows, F, stats, g, b);
}

extern "C" void kernel_launch(void* const* d_in, const int* in_sizes, int n_in,
                              void* d_out, int out_size, void* d_ws, size_t ws_size,
                              hipStream_t stream) {
    (void)in_sizes; (void)n_in; (void)out_size; (void)ws_size;

    const float* x          = (const float*)d_in[0];
    const int*   edge_index = (const int*)  d_in[1];
    const int*   batch      = (const int*)  d_in[2];
    const int*   perm       = (const int*)  d_in[3];
    const int*   row = edge_index;
    const int*   col = edge_index + N_EDGES;

    int p = 4;
    const float* bn_feat_g   = (const float*)d_in[p++];
    const float* bn_feat_b   = (const float*)d_in[p++];
    const float* conv_feat_w = (const float*)d_in[p++];
    const float* gin_w1      = (const float*)d_in[p++];
    const float* gin_b1      = (const float*)d_in[p++];
    const float* gin_bn_g    = (const float*)d_in[p++];
    const float* gin_bn_b    = (const float*)d_in[p++];
    const float* gin_w2      = (const float*)d_in[p++];
    const float* gin_b2      = (const float*)d_in[p++];
    const float* edge_att_w  = (const float*)d_in[p++];
    const float* edge_att_b  = (const float*)d_in[p++];
    const float* node_att_w  = (const float*)d_in[p++];
    const float* node_att_b  = (const float*)d_in[p++];
    const float* bnc_g       = (const float*)d_in[p++];
    const float* bnc_b       = (const float*)d_in[p++];
    const float* bno_g       = (const float*)d_in[p++];
    const float* bno_b       = (const float*)d_in[p++];
    const float* ctx_w       = (const float*)d_in[p++];
    const float* ctx_b       = (const float*)d_in[p++];
    const float* obj_w       = (const float*)d_in[p++];
    const float* obj_b       = (const float*)d_in[p++];
    const float* fc1_bn_c_g  = (const float*)d_in[p++];
    const float* fc1_bn_c_b  = (const float*)d_in[p++];
    const float* fc1_c_w     = (const float*)d_in[p++];
    const float* fc1_c_b     = (const float*)d_in[p++];
    const float* fc2_bn_c_g  = (const float*)d_in[p++];
    const float* fc2_bn_c_b  = (const float*)d_in[p++];
    const float* fc2_c_w     = (const float*)d_in[p++];
    const float* fc2_c_b     = (const float*)d_in[p++];
    const float* fc1_bn_o_g  = (const float*)d_in[p++];
    const float* fc1_bn_o_b  = (const float*)d_in[p++];
    const float* fc1_o_w     = (const float*)d_in[p++];
    const float* fc1_o_b     = (const float*)d_in[p++];
    const float* fc2_bn_o_g  = (const float*)d_in[p++];
    const float* fc2_bn_o_b  = (const float*)d_in[p++];
    const float* fc2_o_w     = (const float*)d_in[p++];
    const float* fc2_o_b     = (const float*)d_in[p++];
    const float* fc1_bn_co_g = (const float*)d_in[p++];
    const float* fc1_bn_co_b = (const float*)d_in[p++];
    const float* fc1_co_w    = (const float*)d_in[p++];
    const float* fc1_co_b    = (const float*)d_in[p++];
    const float* fc2_bn_co_g = (const float*)d_in[p++];
    const float* fc2_bn_co_b = (const float*)d_in[p++];
    const float* fc2_co_w    = (const float*)d_in[p++];
    const float* fc2_co_b    = (const float*)d_in[p++];

    float* out = (float*)d_out;

    // ---- workspace layout
    float* ws = (float*)d_ws;
    size_t off = 0;
    const size_t NB = (size_t)N_NODES * HDIM;
    auto alloc = [&](size_t n) { float* q = ws + off; off += n; return q; };
    float* bufA = alloc(NB);           // h
    float* bufB = alloc(NB);           // z / xc / out_c
    float* bufC = alloc(NB);           // t / xo / out_o
    float* bufD = alloc(NB);           // vw
    float* ewc  = alloc(N_EDGES);
    float* ewo  = alloc(N_EDGES);
    float* deg  = alloc(N_NODES);
    float* stats= alloc(512);
    float* xcg  = alloc((size_t)NGRAPH * HDIM);
    float* xog  = alloc((size_t)NGRAPH * HDIM);
    float* t1   = alloc((size_t)NGRAPH * 256);
    float* t2   = alloc((size_t)NGRAPH * HDIM);
    float* t3   = alloc((size_t)NGRAPH * HDIM);

    const int gemmGridN = (N_NODES + 63) / 64;   // 782
    const int gemmGridG = (NGRAPH + 63) / 64;    // 8

    // ---- Stage 1: h = relu(bn(x) @ conv_feat_w)
    run_bn(x, bufC, N_NODES, 128, bn_feat_g, bn_feat_b, false, stats, stream);
    gemm128_kernel<false, false, true><<<gemmGridN, 256, 0, stream>>>(
        bufC, 128, conv_feat_w, nullptr, bufA, N_NODES);

    // ---- Stage 2: GIN layers
    for (int l = 0; l < NLAYERS; ++l) {
        copy_kernel<<<4096, 256, 0, stream>>>(bufB, bufA, (long)NB);   // z = h
        gin_scatter_kernel<<<4096, 128, 0, stream>>>(bufA, row, col, bufB, N_EDGES); // z += agg
        gemm128_kernel<false, true, false><<<gemmGridN, 256, 0, stream>>>(
            bufB, 128, gin_w1 + (size_t)l * HDIM * HDIM, gin_b1 + l * HDIM, bufC, N_NODES);
        run_bn(bufC, bufC, N_NODES, 128, gin_bn_g + l * HDIM, gin_bn_b + l * HDIM, true, stats, stream);
        gemm128_kernel<false, true, true><<<gemmGridN, 256, 0, stream>>>(
            bufC, 128, gin_w2 + (size_t)l * HDIM * HDIM, gin_b2 + l * HDIM, bufA, N_NODES);
    }

    // ---- Stage 3: attentions
    edge_att_kernel<<<4096, 128, 0, stream>>>(bufA, row, col, edge_att_w, edge_att_b,
                                              ewc, ewo, N_EDGES);
    node_att_kernel<<<4096, 128, 0, stream>>>(bufA, node_att_w, node_att_b,
                                              bufB, bufC, N_NODES);   // xc -> B, xo -> C

    // ---- Stage 4: context branch (xc in bufB)
    run_bn(bufB, bufB, N_NODES, 128, bnc_g, bnc_b, false, stats, stream);
    gemm128_kernel<false, false, false><<<gemmGridN, 256, 0, stream>>>(
        bufB, 128, ctx_w, nullptr, bufD, N_NODES);                    // vw_c
    zero_kernel<<<256, 256, 0, stream>>>(deg, N_NODES);
    deg_kernel<<<2048, 256, 0, stream>>>(ewc, row, deg, N_EDGES);
    dinv_kernel<<<256, 256, 0, stream>>>(deg, N_NODES);
    selfloop_kernel<<<4096, 256, 0, stream>>>(bufD, deg, bufB, (long)NB);
    gcn_scatter_kernel<<<4096, 128, 0, stream>>>(bufD, row, col, ewc, deg, bufB, N_EDGES);
    zero_kernel<<<64, 256, 0, stream>>>(xcg, (long)NGRAPH * HDIM);
    pool_kernel<<<4096, 128, 0, stream>>>(bufB, batch, ctx_b, xcg, N_NODES);

    // ---- Stage 5: object branch (xo in bufC)
    run_bn(bufC, bufC, N_NODES, 128, bno_g, bno_b, false, stats, stream);
    gemm128_kernel<false, false, false><<<gemmGridN, 256, 0, stream>>>(
        bufC, 128, obj_w, nullptr, bufD, N_NODES);                    // vw_o
    zero_kernel<<<256, 256, 0, stream>>>(deg, N_NODES);
    deg_kernel<<<2048, 256, 0, stream>>>(ewo, row, deg, N_EDGES);
    dinv_kernel<<<256, 256, 0, stream>>>(deg, N_NODES);
    selfloop_kernel<<<4096, 256, 0, stream>>>(bufD, deg, bufC, (long)NB);
    gcn_scatter_kernel<<<4096, 128, 0, stream>>>(bufD, row, col, ewo, deg, bufC, N_EDGES);
    zero_kernel<<<64, 256, 0, stream>>>(xog, (long)NGRAPH * HDIM);
    pool_kernel<<<4096, 128, 0, stream>>>(bufC, batch, obj_b, xog, N_NODES);

    // ---- Stage 6: heads
    // head c
    run_bn(xcg, t1, NGRAPH, 128, fc1_bn_c_g, fc1_bn_c_b, false, stats, stream);
    gemm128_kernel<false, true, true><<<gemmGridG, 256, 0, stream>>>(
        t1, 128, fc1_c_w, fc1_c_b, t2, NGRAPH);
    run_bn(t2, t3, NGRAPH, 128, fc2_bn_c_g, fc2_bn_c_b, false, stats, stream);
    fc2_lsm_kernel<<<NGRAPH, 128, 0, stream>>>(t3, fc2_c_w, fc2_c_b, out, NGRAPH);

    // head o
    run_bn(xog, t1, NGRAPH, 128, fc1_bn_o_g, fc1_bn_o_b, false, stats, stream);
    gemm128_kernel<false, true, true><<<gemmGridG, 256, 0, stream>>>(
        t1, 128, fc1_o_w, fc1_o_b, t2, NGRAPH);
    run_bn(t2, t3, NGRAPH, 128, fc2_bn_o_g, fc2_bn_o_b, false, stats, stream);
    fc2_lsm_kernel<<<NGRAPH, 128, 0, stream>>>(t3, fc2_o_w, fc2_o_b, out + NGRAPH * NCLS, NGRAPH);

    // head co: xco = [xc[perm], xo]  (500 x 256), K=256 GEMM = two accumulating K=128 GEMMs
    concat_perm_kernel<<<512, 256, 0, stream>>>(xcg, xog, perm, t1);
    run_bn(t1, t1, NGRAPH, 256, fc1_bn_co_g, fc1_bn_co_b, false, stats, stream);
    gemm128_kernel<false, false, false><<<gemmGridG, 256, 0, stream>>>(
        t1, 256, fc1_co_w, nullptr, t2, NGRAPH);                            // cols 0..127
    gemm128_kernel<true, true, true><<<gemmGridG, 256, 0, stream>>>(
        t1 + 128, 256, fc1_co_w + 128 * 128, fc1_co_b, t2, NGRAPH);         // cols 128..255
    run_bn(t2, t3, NGRAPH, 128, fc2_bn_co_g, fc2_bn_co_b, false, stats, stream);
    fc2_lsm_kernel<<<NGRAPH, 128, 0, stream>>>(t3, fc2_co_w, fc2_co_b,
                                               out + 2 * NGRAPH * NCLS, NGRAPH);
}
